// Encoder_56805237456949
// MI455X (gfx1250) — matrix operand order repaired and checked by reading, not compiled, not run on hardware
//
#include <hip/hip_runtime.h>
#include <hip/hip_bf16.h>

// ---- problem dims ----
#define NB  16      // batch
#define NS  1024    // seq len
#define NF  64      // input feat
#define ND  512     // model dim
#define NH  8       // heads
#define NDH 64      // head dim
#define NL  6       // layers
#define NM  (NB*NS) // 16384 token rows

typedef __attribute__((ext_vector_type(16))) __bf16 bf16x16;
typedef __attribute__((ext_vector_type(8)))  float  f32x8;
typedef __attribute__((ext_vector_type(4)))  unsigned int u32x4;
typedef __attribute__((ext_vector_type(8)))  int i32x8;
typedef __attribute__((ext_vector_type(4)))  int i32x4;

union Frag {
  uint4 u[2];
  bf16x16 v;
  unsigned short s[16];
};

__device__ __forceinline__ unsigned short f2bf(float f) {
  unsigned int u = __float_as_uint(f);
  u += 0x7fffu + ((u >> 16) & 1u);           // round-to-nearest-even
  return (unsigned short)(u >> 16);
}

__device__ __forceinline__ f32x8 wmma_bf16(const Frag& a, const Frag& b, f32x8 c) {
  // (neg_a, A, neg_b, B, c_mod, C, reuse_a, reuse_b)
  return __builtin_amdgcn_wmma_f32_16x16x32_bf16(false, a.v, false, b.v, (short)0, c, false, false);
}

// TDM: DMA a [64 rows x 32 cols] bf16 tile (row stride 512 elems) Global->LDS.
// D# per CDNA5 ISA ch8: group0 = {flags, lds_addr, global_addr(57b)+type},
// group1 = {mask/data_size, tensor dims/tile dims/strides}. 2-D tile => the
// remaining groups are unused (zero). Tracked by TENSORcnt.
__device__ __forceinline__ void tdm_load_tile_64x32(unsigned lds_addr,
                                                    const unsigned short* gsrc) {
  unsigned long long ga = (unsigned long long)(uintptr_t)gsrc;
  u32x4 g0;
  g0[0] = 1u;                                            // count=1, user D#
  g0[1] = lds_addr;                                      // LDS byte address
  g0[2] = (unsigned)(ga & 0xFFFFFFFFu);                  // global addr [31:0]
  g0[3] = (unsigned)((ga >> 32) & 0x1FFFFFFu)            // global addr [56:32]
          | 0x80000000u;                                 // type = 2 ("image")
  i32x8 g1;
  g1[0] = 1 << 16;                                       // data_size=2B, wg_mask=0
  g1[1] = (int)(32u << 16);                              // tensor_dim0[15:0]=32
  g1[2] = (int)(64u << 16);                              // dim0 hi=0 | tensor_dim1 lo=64
  g1[3] = (int)(32u << 16);                              // dim1 hi=0 | tile_dim0=32
  g1[4] = (int)(64u);                                    // tile_dim1=64, tile_dim2=0
  g1[5] = (int)(512u);                                   // tensor_dim0_stride=512 (lo32)
  g1[6] = 0;                                             // stride hi | dim1_stride lo
  g1[7] = 0;
  i32x4 z4 = {0, 0, 0, 0};
  i32x8 z8 = {0, 0, 0, 0, 0, 0, 0, 0};
  // 6-arg (clang-23 / therock-10.0) form
  __builtin_amdgcn_tensor_load_to_lds(g0, g1, z4, z4, z8, 0);
}

// ---------------------------------------------------------------------------
// Weight conversion: f32 [L,D,D] (k-major) -> bf16 [L,D,D] transposed (n-major)
// ---------------------------------------------------------------------------
__global__ __launch_bounds__(256) void k_wt(
    const float* __restrict__ Wq, const float* __restrict__ Wk,
    const float* __restrict__ Wv, const float* __restrict__ Wo,
    unsigned short* __restrict__ Tq, unsigned short* __restrict__ Tk,
    unsigned short* __restrict__ Tv, unsigned short* __restrict__ To) {
  int idx = blockIdx.x * blockDim.x + threadIdx.x;   // l*D*D + n*D + k
  int l  = idx >> 18;                                // D*D = 2^18
  int nk = idx & 262143;
  int n = nk >> 9, k = nk & 511;
  int src = (l << 18) + (k << 9) + n;
  Tq[idx] = f2bf(Wq[src]);
  Tk[idx] = f2bf(Wk[src]);
  Tv[idx] = f2bf(Wv[src]);
  To[idx] = f2bf(Wo[src]);
}

// ---------------------------------------------------------------------------
// Embedding: h = (x @ Wemb + bemb) * sqrt(D) + posenc ; also bf16 copy
// ---------------------------------------------------------------------------
__global__ __launch_bounds__(256) void k_embed(
    const float* __restrict__ x, const float* __restrict__ Wemb,
    const float* __restrict__ bemb, float* __restrict__ h,
    unsigned short* __restrict__ hb) {
  int idx = blockIdx.x * blockDim.x + threadIdx.x;   // < NM*ND
  int m = idx >> 9;
  int n = idx & 511;
  const float* xr = x + m * NF;
  float acc = bemb[n];
#pragma unroll
  for (int f = 0; f < NF; ++f) acc = fmaf(xr[f], Wemb[f * ND + n], acc);
  int s = m & (NS - 1);
  float ip = (float)(n & ~1);
  float angle = (float)s * __expf(ip * (-9.210340371976184f / 512.0f)); // 10000^(-ip/512)
  float pe = (n & 1) ? __cosf(angle) : __sinf(angle);
  float val = acc * 22.627416997969522f + pe;        // sqrt(512)
  h[idx]  = val;
  hb[idx] = f2bf(val);
}

// ---------------------------------------------------------------------------
// bf16 WMMA GEMM: C[M=16384, N=512] = A[M,512] x WT[N,512]^T (+bias, epilogue)
//   mode 0/1: Q/K proj -> bf16 [B,H,S,DH]  (mode 0 scaled by 1/sqrt(DH))
//   mode 2  : V proj   -> bf16 [B,H,DH,S]  (transposed for PV B-fragments)
//   mode 3  : O proj   -> f32 tmp = gemm + bias + residual(h)
// Block: 256 thr (8 waves). Tile 128(M) x 64(N). Weight tiles staged to LDS
// by the Tensor Data Mover (wave 0 issues, TENSORcnt), double-buffered so the
// TDM transfer for step i+1 overlaps the WMMAs of step i. One barrier/step.
// ---------------------------------------------------------------------------
__global__ __launch_bounds__(256) void k_gemm(
    const unsigned short* __restrict__ A,   // bf16 [16384,512]
    const unsigned short* __restrict__ WT,  // bf16 [512,512] n-major
    const float* __restrict__ bias,         // [512]
    const float* __restrict__ resid,        // mode 3 only
    float scale, int mode,
    unsigned short* __restrict__ outb,      // modes 0-2
    float* __restrict__ outf) {             // mode 3
  __shared__ unsigned short sB[2][64 * 32];

  const int tid  = threadIdx.x;
  const int lane = tid & 31;
  const int w    = tid >> 5;
  const int ln   = lane & 15;
  const int hi   = lane >> 4;                  // 0 | 1
  const int mb   = blockIdx.x * 128 + w * 16;  // wave's 16 rows
  const int nb   = blockIdx.y * 64;

  f32x8 acc[4] = {};

  const unsigned short* aptr  = A + (mb + ln) * ND + hi * 16;
  const unsigned short* wbase = WT + nb * ND;              // tile row 0, k 0
  const unsigned lds0 = (unsigned)(uintptr_t)(void*)&sB[0][0];
  const unsigned lds1 = (unsigned)(uintptr_t)(void*)&sB[1][0];

  // prologue: DMA first weight tile into buffer 0
  if (w == 0) {
    tdm_load_tile_64x32(lds0, wbase);
    __builtin_amdgcn_s_wait_tensorcnt(0);
  }
  __syncthreads();

  for (int it = 0; it < ND / 32; ++it) {
    const int k0  = it * 32;
    const int cur = it & 1;
    // kick off DMA of the next tile into the other buffer
    if (w == 0 && it + 1 < ND / 32)
      tdm_load_tile_64x32(cur ? lds0 : lds1, wbase + k0 + 32);

    Frag a;
    a.u[0] = *(const uint4*)(aptr + k0);
    a.u[1] = *(const uint4*)(aptr + k0 + 8);
    if (k0 + 32 < ND) __builtin_prefetch(aptr + k0 + 32, 0, 3);

    const unsigned short* bb = sB[cur] + ln * 32 + hi * 16;
    Frag b0, b1, b2, b3;
    b0.u[0] = *(const uint4*)(bb);            b0.u[1] = *(const uint4*)(bb + 8);
    b1.u[0] = *(const uint4*)(bb + 512);      b1.u[1] = *(const uint4*)(bb + 520);
    b2.u[0] = *(const uint4*)(bb + 1024);     b2.u[1] = *(const uint4*)(bb + 1032);
    b3.u[0] = *(const uint4*)(bb + 1536);     b3.u[1] = *(const uint4*)(bb + 1544);

    acc[0] = wmma_bf16(a, b0, acc[0]);
    acc[1] = wmma_bf16(a, b1, acc[1]);
    acc[2] = wmma_bf16(a, b2, acc[2]);
    acc[3] = wmma_bf16(a, b3, acc[3]);

    // next buffer must be complete (wave 0) and everyone done reading `cur`
    if (w == 0 && it + 1 < ND / 32)
      __builtin_amdgcn_s_wait_tensorcnt(0);
    __syncthreads();
  }

  // epilogue
#pragma unroll
  for (int t = 0; t < 4; ++t) {
    int n = nb + t * 16 + ln;
    float bn = bias[n];
#pragma unroll
    for (int r = 0; r < 8; ++r) {
      int m = mb + r + hi * 8;
      float val = acc[t][r] + bn;
      if (mode == 3) {
        int idx = m * ND + n;
        outf[idx] = val + resid[idx];
      } else {
        val *= scale;
        int b = m >> 10, s = m & 1023;
        int hh = n >> 6, d = n & 63;
        int idx = (mode == 2) ? ((b * NH + hh) * NDH + d) * NS + s
                              : ((b * NH + hh) * NS + s) * NDH + d;
        outb[idx] = f2bf(val);
      }
    }
  }
}

// ---------------------------------------------------------------------------
// Flash attention per (b,h): S^T = K x Q^T (C-layout == A-layout of P for PV),
// online softmax, acc += P x V^T. Block = 128 thr (4 waves x 16 queries).
// ---------------------------------------------------------------------------
__global__ __launch_bounds__(128) void k_attn(
    const unsigned short* __restrict__ q,    // bf16 [B,H,S,DH], pre-scaled
    const unsigned short* __restrict__ k,    // bf16 [B,H,S,DH]
    const unsigned short* __restrict__ vT,   // bf16 [B,H,DH,S]
    const float* __restrict__ mask,          // [B,S]
    unsigned short* __restrict__ attnb) {    // bf16 [B,S,H*DH]
  __shared__ float smask[NS];

  const int lane = threadIdx.x & 31;
  const int w    = threadIdx.x >> 5;
  const int ln   = lane & 15;
  const int hi   = lane >> 4;
  const int bh   = blockIdx.x;               // b*NH + h
  const int b    = bh >> 3;
  const int hd   = bh & 7;
  const int qb   = blockIdx.y * 64 + w * 16; // this wave's 16 query rows

  for (int i = threadIdx.x; i < NS; i += 128) smask[i] = mask[b * NS + i] * (-1e9f);
  __syncthreads();

  // Q fragments (B-operand of K x Q^T), dh chunks 0/1
  Frag qf0, qf1;
  {
    const unsigned short* qp = q + ((size_t)bh * NS + qb + ln) * NDH + hi * 8;
    qf0.u[0] = *(const uint4*)(qp);      qf0.u[1] = *(const uint4*)(qp + 16);
    qf1.u[0] = *(const uint4*)(qp + 32); qf1.u[1] = *(const uint4*)(qp + 48);
  }

  f32x8 acc[4] = {};
  float mrun = -1e30f, lrun = 0.f;
  const unsigned short* kbase = k  + (size_t)bh * NS * NDH;
  const unsigned short* vbase = vT + (size_t)bh * NDH * NS;

  for (int kb0 = 0; kb0 < NS; kb0 += 32) {
    // score tiles S^T: rows=keys (A), cols=queries (B)
    f32x8 st0 = {}, st1 = {};
    {
      const unsigned short* kp0 = kbase + (size_t)(kb0 + ln) * NDH + hi * 8;
      Frag ka;
      ka.u[0] = *(const uint4*)(kp0);      ka.u[1] = *(const uint4*)(kp0 + 16);
      st0 = wmma_bf16(ka, qf0, st0);
      ka.u[0] = *(const uint4*)(kp0 + 32); ka.u[1] = *(const uint4*)(kp0 + 48);
      st0 = wmma_bf16(ka, qf1, st0);
      const unsigned short* kp1 = kp0 + 16 * NDH;
      ka.u[0] = *(const uint4*)(kp1);      ka.u[1] = *(const uint4*)(kp1 + 16);
      st1 = wmma_bf16(ka, qf0, st1);
      ka.u[0] = *(const uint4*)(kp1 + 32); ka.u[1] = *(const uint4*)(kp1 + 48);
      st1 = wmma_bf16(ka, qf1, st1);
    }

    // mask + online softmax (stats per query = per lane-pair {l, l+16})
    float mx = -1e30f;
#pragma unroll
    for (int r = 0; r < 8; ++r) {
      st0[r] += smask[kb0 + r + hi * 8];
      st1[r] += smask[kb0 + 16 + r + hi * 8];
      mx = fmaxf(mx, fmaxf(st0[r], st1[r]));
    }
    mx = fmaxf(mx, __shfl_xor(mx, 16, 32));
    float mnew = fmaxf(mrun, mx);
    float fac  = __expf(mrun - mnew);
    float ps   = 0.f;
    Frag pa;
#pragma unroll
    for (int r = 0; r < 8; ++r) {
      float p0 = __expf(st0[r] - mnew);
      float p1 = __expf(st1[r] - mnew);
      ps += p0 + p1;
      pa.s[r]     = f2bf(p0);
      pa.s[r + 8] = f2bf(p1);
    }
    ps += __shfl_xor(ps, 16, 32);
    lrun = lrun * fac + ps;
    mrun = mnew;

    // rescale accumulators (row q lives in VGPR r; factor lives in lane q%16)
#pragma unroll
    for (int r = 0; r < 8; ++r) {
      float fr = __shfl(fac, r + hi * 8, 32);
      acc[0][r] *= fr; acc[1][r] *= fr; acc[2][r] *= fr; acc[3][r] *= fr;
    }

    // PV: acc[t] += P(16x32) x V^T-column tiles (K=32 keys)
#pragma unroll
    for (int t = 0; t < 4; ++t) {
      const unsigned short* vp = vbase + (size_t)(t * 16 + ln) * NS + kb0 + hi * 8;
      Frag vb;
      vb.u[0] = *(const uint4*)(vp);
      vb.u[1] = *(const uint4*)(vp + 16);
      acc[t] = wmma_bf16(pa, vb, acc[t]);
    }
  }

  float inv = 1.0f / lrun;
#pragma unroll
  for (int r = 0; r < 8; ++r) {
    float ri = __shfl(inv, r + hi * 8, 32);
    int qrow = qb + r + hi * 8;
    unsigned short* op = attnb + ((size_t)(b * NS + qrow)) * ND + hd * NDH;
#pragma unroll
    for (int t = 0; t < 4; ++t)
      op[t * 16 + ln] = f2bf(acc[t][r] * ri);
  }
}

// ---------------------------------------------------------------------------
// LayerNorm: one wave per row of 512. Writes f32 result + bf16 copy.
// ---------------------------------------------------------------------------
__global__ __launch_bounds__(256) void k_ln(
    const float* __restrict__ tin, const float* __restrict__ gamma,
    const float* __restrict__ beta, float* __restrict__ hout,
    unsigned short* __restrict__ hb) {
  int row  = blockIdx.x * 8 + (threadIdx.x >> 5);
  int lane = threadIdx.x & 31;
  const float* rp = tin + (size_t)row * ND;
  float4 v[4];
  float s = 0.f;
#pragma unroll
  for (int i = 0; i < 4; ++i) {
    v[i] = *(const float4*)(rp + lane * 4 + i * 128);
    s += v[i].x + v[i].y + v[i].z + v[i].w;
  }
#pragma unroll
  for (int m = 1; m < 32; m <<= 1) s += __shfl_xor(s, m, 32);
  float mean = s * (1.0f / ND);
  float qv = 0.f;
#pragma unroll
  for (int i = 0; i < 4; ++i) {
    float a = v[i].x - mean, c = v[i].y - mean, d = v[i].z - mean, e = v[i].w - mean;
    qv += a * a + c * c + d * d + e * e;
  }
#pragma unroll
  for (int m = 1; m < 32; m <<= 1) qv += __shfl_xor(qv, m, 32);
  float rstd = rsqrtf(qv * (1.0f / ND) + 1e-6f);
#pragma unroll
  for (int i = 0; i < 4; ++i) {
    int c0 = lane * 4 + i * 128;
    float o0 = (v[i].x - mean) * rstd * gamma[c0 + 0] + beta[c0 + 0];
    float o1 = (v[i].y - mean) * rstd * gamma[c0 + 1] + beta[c0 + 1];
    float o2 = (v[i].z - mean) * rstd * gamma[c0 + 2] + beta[c0 + 2];
    float o3 = (v[i].w - mean) * rstd * gamma[c0 + 3] + beta[c0 + 3];
    float4 ov = {o0, o1, o2, o3};
    *(float4*)(hout + (size_t)row * ND + c0) = ov;
    unsigned short* hp = hb + (size_t)row * ND + c0;
    hp[0] = f2bf(o0); hp[1] = f2bf(o1); hp[2] = f2bf(o2); hp[3] = f2bf(o3);
  }
}

// ---------------------------------------------------------------------------
extern "C" void kernel_launch(void* const* d_in, const int* in_sizes, int n_in,
                              void* d_out, int out_size, void* d_ws, size_t ws_size,
                              hipStream_t stream) {
  const float* x     = (const float*)d_in[0];
  const float* mask  = (const float*)d_in[1];
  const float* Wemb  = (const float*)d_in[3];
  const float* bemb  = (const float*)d_in[4];
  const float* Wq    = (const float*)d_in[5];
  const float* bq    = (const float*)d_in[6];
  const float* Wk    = (const float*)d_in[7];
  const float* bk    = (const float*)d_in[8];
  const float* Wv    = (const float*)d_in[9];
  const float* bv    = (const float*)d_in[10];
  const float* Wo    = (const float*)d_in[11];
  const float* bo    = (const float*)d_in[12];
  const float* gamma = (const float*)d_in[13];
  const float* beta  = (const float*)d_in[14];

  char* ws = (char*)d_ws;
  size_t off = 0;
  auto alloc = [&](size_t bytes) -> void* {
    void* p = ws + off;
    off += (bytes + 255) & ~(size_t)255;
    return p;
  };
  float*          h     = (float*)alloc((size_t)NM * ND * 4);
  float*          tmp   = (float*)alloc((size_t)NM * ND * 4);
  unsigned short* hb    = (unsigned short*)alloc((size_t)NM * ND * 2);
  unsigned short* attnb = (unsigned short*)alloc((size_t)NM * ND * 2);
  unsigned short* qbuf  = (unsigned short*)alloc((size_t)NM * ND * 2);
  unsigned short* kbuf  = (unsigned short*)alloc((size_t)NM * ND * 2);
  unsigned short* vtbuf = (unsigned short*)alloc((size_t)NM * ND * 2);
  unsigned short* Tq    = (unsigned short*)alloc((size_t)NL * ND * ND * 2);
  unsigned short* Tk    = (unsigned short*)alloc((size_t)NL * ND * ND * 2);
  unsigned short* Tv    = (unsigned short*)alloc((size_t)NL * ND * ND * 2);
  unsigned short* To    = (unsigned short*)alloc((size_t)NL * ND * ND * 2);

  k_wt<<<(NL * ND * ND) / 256, 256, 0, stream>>>(Wq, Wk, Wv, Wo, Tq, Tk, Tv, To);
  k_embed<<<(NM * ND) / 256, 256, 0, stream>>>(x, Wemb, bemb, h, hb);

  const dim3 gGemm(NM / 128, ND / 64);
  const dim3 gAttn(NB * NH, NS / 64);
  const float qscale = 0.125f;  // 1/sqrt(DH)

  for (int l = 0; l < NL; ++l) {
    size_t wo = (size_t)l * ND * ND;
    size_t vo = (size_t)l * ND;
    k_gemm<<<gGemm, 256, 0, stream>>>(hb, Tq + wo, bq + vo, nullptr, qscale, 0, qbuf, nullptr);
    k_gemm<<<gGemm, 256, 0, stream>>>(hb, Tk + wo, bk + vo, nullptr, 1.0f,   1, kbuf, nullptr);
    k_gemm<<<gGemm, 256, 0, stream>>>(hb, Tv + wo, bv + vo, nullptr, 1.0f,   2, vtbuf, nullptr);
    k_attn<<<gAttn, 128, 0, stream>>>(qbuf, kbuf, vtbuf, mask, attnb);
    k_gemm<<<gGemm, 256, 0, stream>>>(attnb, To + wo, bo + vo, h, 1.0f, 3, nullptr, tmp);
    float* dst = (l == NL - 1) ? (float*)d_out : h;
    k_ln<<<NM / 8, 256, 0, stream>>>(tmp, gamma + vo, beta + vo, dst, hb);
  }
}
